// RGA_direct_encode_whole_15401752723469
// MI455X (gfx1250) — compile-verified, matrix operand order checked by reading, not run on hardware
//
#include <hip/hip_runtime.h>
#include <hip/hip_bf16.h>

typedef __attribute__((ext_vector_type(16))) __bf16 v16bf;
typedef __attribute__((ext_vector_type(8)))  float  v8f;

__device__ __forceinline__ unsigned flipf(float f) {
  unsigned u = __float_as_uint(f);
  return (u & 0x80000000u) ? ~u : (u | 0x80000000u);
}
__device__ __forceinline__ float unflipf(unsigned u) {
  u = (u & 0x80000000u) ? (u & 0x7fffffffu) : ~u;
  return __uint_as_float(u);
}

// ISA 7.12.2: 16-bit A-matrix 16x32 fragment K index for (half, element e)
__device__ __forceinline__ int kperm(int half, int e) {
  int v = e >> 1, p = e & 1;
  return (v < 4) ? (half * 8 + v * 2 + p) : (16 + half * 8 + (v - 4) * 2 + p);
}

// ---------------- generic fills ----------------
__global__ void fill_f32_kernel(float* p, float v, size_t n) {
  size_t i = (size_t)blockIdx.x * blockDim.x + threadIdx.x;
  if (i < n) p[i] = v;
}
__global__ void fill_u32_kernel(unsigned* p, unsigned v, size_t n) {
  size_t i = (size_t)blockIdx.x * blockDim.x + threadIdx.x;
  if (i < n) p[i] = v;
}

// ---------------- self-loop edge_attr ----------------
__global__ void loop_accum_kernel(const int* __restrict__ dst, const float* __restrict__ ea,
                                  float* cnt, float* easum, int E) {
  int e = blockIdx.x * blockDim.x + threadIdx.x;
  if (e >= E) return;
  int d = dst[e];
  atomicAdd(&cnt[d], 1.0f);
  #pragma unroll
  for (int k = 0; k < 16; ++k) atomicAdd(&easum[(size_t)d * 16 + k], ea[(size_t)e * 16 + k]);
}

__global__ void build_edges_kernel(const int* __restrict__ src, const int* __restrict__ dst,
                                   const float* __restrict__ ea, const float* __restrict__ cnt,
                                   const float* __restrict__ easum,
                                   int* src2, int* dst2, float* ea2, int E, int Nn) {
  int i = blockIdx.x * blockDim.x + threadIdx.x;
  if (i < E) {
    src2[i] = src[i]; dst2[i] = dst[i];
    #pragma unroll
    for (int k = 0; k < 16; ++k) ea2[(size_t)i * 16 + k] = ea[(size_t)i * 16 + k];
  } else if (i < E + Nn) {
    int nid = i - E;
    src2[i] = nid; dst2[i] = nid;
    float c = fmaxf(cnt[nid], 1.0f);
    #pragma unroll
    for (int k = 0; k < 16; ++k) ea2[(size_t)i * 16 + k] = easum[(size_t)nid * 16 + k] / c;
  }
}

// ---------------- BatchNorm (training-mode batch stats) ----------------
__global__ __launch_bounds__(256) void bn_stats_kernel(const float* __restrict__ X, int M, int C,
                                                       float* mu, float* rstd) {
  int c = blockIdx.x;
  __shared__ float s1[256], s2[256];
  float a = 0.f, b = 0.f;
  for (int r = threadIdx.x; r < M; r += 256) {
    float v = X[(size_t)r * C + c];
    a += v; b += v * v;
  }
  s1[threadIdx.x] = a; s2[threadIdx.x] = b;
  __syncthreads();
  for (int o = 128; o > 0; o >>= 1) {
    if ((int)threadIdx.x < o) { s1[threadIdx.x] += s1[threadIdx.x + o]; s2[threadIdx.x] += s2[threadIdx.x + o]; }
    __syncthreads();
  }
  if (threadIdx.x == 0) {
    float m = s1[0] / (float)M;
    float var = s2[0] / (float)M - m * m;
    mu[c] = m;
    rstd[c] = rsqrtf(var + 1e-5f);
  }
}

__global__ void bn_apply_kernel(const float* __restrict__ X, const float* __restrict__ res,
                                const float* __restrict__ mu, const float* __restrict__ rstd,
                                const float* __restrict__ g, const float* __restrict__ be,
                                float* __restrict__ Y, int M, int C, int relu) {
  size_t i = (size_t)blockIdx.x * blockDim.x + threadIdx.x;
  if (i >= (size_t)M * C) return;
  int c = (int)(i % C);
  float y = (X[i] - mu[c]) * rstd[c] * g[c] + be[c];
  if (res) y += res[i];
  if (relu) y = fmaxf(y, 0.f);
  Y[i] = y;
}

// ---------------- fragment repack: fp32 matrix -> bf16 WMMA fragment layout -------------
// Produces tiles of 512 bf16: tile (rt, kt), lane l owns 16 contiguous bf16 such that a
// single 32B load yields the v16bf WMMA operand. A-matrix: element (r,k) = src[r*rs + k*ks].
// Zero-pads r >= R or k >= K so the GEMM inner loop needs no bounds checks.
__global__ __launch_bounds__(256) void pack_frag_kernel(
    const float* __restrict__ src, __bf16* __restrict__ dst,
    int R, int K, int rs, int ks, int Kt, size_t total) {
  size_t t = (size_t)blockIdx.x * blockDim.x + threadIdx.x;
  if (t >= total) return;
  int lane = (int)(t & 31);
  size_t tile = t >> 5;
  int kt = (int)(tile % Kt);
  int rt = (int)(tile / Kt);
  int r = rt * 16 + (lane & 15);
  int half = lane >> 4;
  int k0 = kt * 32;
  v16bf out;
  #pragma unroll
  for (int e = 0; e < 16; ++e) {
    int k = k0 + kperm(half, e);
    float v = (r < R && k < K) ? src[(size_t)r * rs + (size_t)k * ks] : 0.f;
    out[e] = (__bf16)v;
  }
  *(v16bf*)(dst + tile * 512 + (size_t)lane * 16) = out;
}

// ---------------- WMMA bf16 GEMM on pre-packed fragments ----------------
// Block = 256 threads = 8 waves as 2 (M) x 4 (N); each wave computes a 32x32 C tile
// (2 A-frags x 2 B-frags -> 4 v_wmma per k-step). No LDS, no barriers.
__device__ __forceinline__ void store_tile(v8f c, int mt, int nt, float* __restrict__ C,
                                           const float* __restrict__ bias, int M, int N, int lane) {
  int col = nt * 16 + (lane & 15);
  if (col >= N) return;
  float bv = bias ? bias[col] : 0.f;
  int half = lane >> 4;
  #pragma unroll
  for (int r = 0; r < 8; ++r) {
    int row = mt * 16 + half * 8 + r;
    if (row < M) C[(size_t)row * N + col] = c[r] + bv;
  }
}

__global__ __launch_bounds__(256) void wmma_gemm_kernel(
    const __bf16* __restrict__ Apack, const __bf16* __restrict__ Bpack,
    const float* __restrict__ bias, float* __restrict__ C,
    int M, int N, int Kt) {
  const int lane = threadIdx.x & 31;
  const int wave = threadIdx.x >> 5;
  const int Mt = (M + 15) >> 4;
  const int Nt = (N + 15) >> 4;
  int mt0 = blockIdx.x * 4 + (wave >> 2) * 2;
  int nt0 = blockIdx.y * 8 + (wave & 3) * 2;
  if (mt0 >= Mt || nt0 >= Nt) return;
  int mt1 = mt0 + 1 < Mt ? mt0 + 1 : mt0;
  int nt1 = nt0 + 1 < Nt ? nt0 + 1 : nt0;

  const v16bf* __restrict__ Av = (const v16bf*)Apack;
  const v16bf* __restrict__ Bv = (const v16bf*)Bpack;
  v8f c00 = {}, c01 = {}, c10 = {}, c11 = {};

  for (int kt = 0; kt < Kt; ++kt) {
    v16bf a0 = Av[((size_t)mt0 * Kt + kt) * 32 + lane];
    v16bf a1 = Av[((size_t)mt1 * Kt + kt) * 32 + lane];
    v16bf b0 = Bv[((size_t)nt0 * Kt + kt) * 32 + lane];
    v16bf b1 = Bv[((size_t)nt1 * Kt + kt) * 32 + lane];
    c00 = __builtin_amdgcn_wmma_f32_16x16x32_bf16(false, a0, false, b0, (short)0, c00, false, false);
    c01 = __builtin_amdgcn_wmma_f32_16x16x32_bf16(false, a0, false, b1, (short)0, c01, false, false);
    c10 = __builtin_amdgcn_wmma_f32_16x16x32_bf16(false, a1, false, b0, (short)0, c10, false, false);
    c11 = __builtin_amdgcn_wmma_f32_16x16x32_bf16(false, a1, false, b1, (short)0, c11, false, false);
  }
  store_tile(c00, mt0, nt0, C, bias, M, N, lane);
  store_tile(c01, mt0, nt1, C, bias, M, N, lane);
  store_tile(c10, mt1, nt0, C, bias, M, N, lane);
  store_tile(c11, mt1, nt1, C, bias, M, N, lane);
}

// ---------------- GATv2 edge passes (wave per (edge, head)) ----------------
__global__ __launch_bounds__(256) void edge_pass1_kernel(
    const int* __restrict__ src2, const int* __restrict__ dst2,
    const float* __restrict__ ea2, const float* __restrict__ We,
    const float* __restrict__ xl, const float* __restrict__ xr,
    const float* __restrict__ att, float* __restrict__ logits,
    unsigned* __restrict__ mmax, int E2, int Hh, int co) {
  int gw = (int)(((size_t)blockIdx.x * blockDim.x + threadIdx.x) >> 5);
  int lane = threadIdx.x & 31;
  if (gw >= E2 * Hh) return;
  int e = gw / Hh, h = gw % Hh;
  int s = src2[e], d = dst2[e];
  int HC = Hh * co;
  float acc = 0.f;
  for (int cc = lane; cc < co; cc += 32) {
    int idx = h * co + cc;
    float eev = 0.f;
    #pragma unroll
    for (int k = 0; k < 16; ++k) eev += ea2[(size_t)e * 16 + k] * We[(size_t)k * HC + idx];
    float sv = xl[(size_t)s * HC + idx] + xr[(size_t)d * HC + idx] + eev;
    sv = sv > 0.f ? sv : 0.2f * sv;            // leaky_relu(0.2)
    acc += sv * att[idx];
  }
  for (int o = 16; o > 0; o >>= 1) acc += __shfl_down(acc, o, 32);
  if (lane == 0) {
    logits[gw] = acc;
    atomicMax(&mmax[(size_t)d * Hh + h], flipf(acc));
  }
}

__global__ void edge_pass2_kernel(const int* __restrict__ dst2, const unsigned* __restrict__ mmax,
                                  float* __restrict__ logits, float* __restrict__ den,
                                  int E2, int Hh) {
  int i = blockIdx.x * blockDim.x + threadIdx.x;
  if (i >= E2 * Hh) return;
  int e = i / Hh, h = i % Hh;
  int d = dst2[e];
  float m = unflipf(mmax[(size_t)d * Hh + h]);
  float ex = __expf(logits[i] - m);
  logits[i] = ex;
  atomicAdd(&den[(size_t)d * Hh + h], ex);
}

__global__ __launch_bounds__(256) void edge_pass3_kernel(
    const int* __restrict__ src2, const int* __restrict__ dst2,
    const float* __restrict__ ex, const float* __restrict__ den,
    const float* __restrict__ xl, float* __restrict__ out,
    int E2, int Hh, int co) {
  int gw = (int)(((size_t)blockIdx.x * blockDim.x + threadIdx.x) >> 5);
  int lane = threadIdx.x & 31;
  if (gw >= E2 * Hh) return;
  int e = gw / Hh, h = gw % Hh;
  int s = src2[e], d = dst2[e];
  float alpha = ex[gw] / den[(size_t)d * Hh + h];
  int HC = Hh * co;
  for (int cc = lane; cc < co; cc += 32)
    atomicAdd(&out[(size_t)d * HC + h * co + cc], alpha * xl[(size_t)s * HC + h * co + cc]);
}

__global__ void head_mean_kernel(const float* __restrict__ acc, const float* __restrict__ b,
                                 float* __restrict__ out, int n, int Hh, int co) {
  size_t i = (size_t)blockIdx.x * blockDim.x + threadIdx.x;
  if (i >= (size_t)n * co) return;
  int r = (int)(i / co), c = (int)(i % co);
  float s = 0.f;
  for (int h = 0; h < Hh; ++h) s += acc[(size_t)r * Hh * co + h * co + c];
  out[i] = s * (1.0f / Hh) + b[c];
}

// ---------------- heads ----------------
__global__ void pair_build_kernel(const float* __restrict__ latent, const int* __restrict__ i0,
                                  const int* __restrict__ i1, float fill,
                                  float* __restrict__ P, int n) {
  int i = blockIdx.x * blockDim.x + threadIdx.x;
  if (i >= n) return;
  int a = i0[i], b = i1[i];
  float* row = P + (size_t)i * 65;
  #pragma unroll
  for (int k = 0; k < 32; ++k) row[k] = latent[(size_t)a * 32 + k];
  #pragma unroll
  for (int k = 0; k < 32; ++k) row[32 + k] = latent[(size_t)b * 32 + k];
  row[64] = fill;
}

__global__ __launch_bounds__(256) void fc6_kernel(const float* __restrict__ U,
                                                  const float* __restrict__ W,
                                                  const float* __restrict__ b,
                                                  float* __restrict__ out, int n) {
  int gw = (int)(((size_t)blockIdx.x * blockDim.x + threadIdx.x) >> 5);
  int lane = threadIdx.x & 31;
  if (gw >= n) return;
  float v = U[(size_t)gw * 32 + lane] * W[lane];
  for (int o = 16; o > 0; o >>= 1) v += __shfl_down(v, o, 32);
  if (lane == 0) out[gw] = fmaxf(v + b[0], 0.f);
}

// =====================================================================
extern "C" void kernel_launch(void* const* d_in, const int* in_sizes, int n_in,
                              void* d_out, int out_size, void* d_ws, size_t ws_size,
                              hipStream_t stream) {
  (void)n_in; (void)out_size; (void)ws_size;
  auto fp = [&](int i) { return (const float*)d_in[i]; };
  auto ip = [&](int i) { return (const int*)d_in[i]; };

  const int Fdim = 128, Hh = 4;
  const int Nn = in_sizes[0] / Fdim;
  const int Ee = in_sizes[1] / 2;
  const int E2 = Ee + Nn;
  const int NB  = in_sizes[3] / 2;
  const int NA  = in_sizes[4] / 2;
  const int ND  = in_sizes[5] / 2;
  const int NH4 = in_sizes[6] / 2;
  int nmax = NB; if (NA > nmax) nmax = NA; if (ND > nmax) nmax = ND; if (NH4 > nmax) nmax = NH4;

  // workspace bump allocator
  char* wsb = (char*)d_ws;
  size_t off = 0;
  auto alloc = [&](size_t bytes) -> void* {
    void* p = wsb + off;
    off += (bytes + 255) & ~(size_t)255;
    return p;
  };
  int*   src2   = (int*)  alloc((size_t)E2 * 4);
  int*   dst2   = (int*)  alloc((size_t)E2 * 4);
  float* cnt    = (float*)alloc((size_t)Nn * 4);
  float* easum  = (float*)alloc((size_t)Nn * 16 * 4);
  float* ea2    = (float*)alloc((size_t)E2 * 16 * 4);
  float* mu     = (float*)alloc(1024 * 4);
  float* rstd   = (float*)alloc(1024 * 4);
  float* xl     = (float*)alloc((size_t)Nn * 1024 * 4);
  float* xr     = (float*)alloc((size_t)Nn * 1024 * 4);
  float* acc    = (float*)alloc((size_t)Nn * 1024 * 4);
  float* gm     = (float*)alloc((size_t)Nn * 256 * 4);
  float* hA     = (float*)alloc((size_t)Nn * 256 * 4);
  float* hB     = (float*)alloc((size_t)Nn * 256 * 4);
  float* logits = (float*)alloc((size_t)E2 * Hh * 4);
  unsigned* mx  = (unsigned*)alloc((size_t)Nn * Hh * 4);
  float* den    = (float*)alloc((size_t)Nn * Hh * 4);
  float* Pbuf   = (float*)alloc((size_t)nmax * 65 * 4);
  float* Ubuf   = (float*)alloc((size_t)nmax * 256 * 4);
  float* Vbuf   = (float*)alloc((size_t)nmax * 256 * 4);
  // packed fragment buffers (bf16): Mt up to ceil(200000/16)=12500, Kt up to 8
  size_t apack_elems = ((size_t)(nmax + 15) / 16) * 8 * 512;
  __bf16* Apack = (__bf16*)alloc(apack_elems * 2);
  __bf16* Bpack = (__bf16*)alloc((size_t)64 * 8 * 512 * 2);

  float* latent = (float*)d_out;            // [Nn,32] first output segment

  auto fillf = [&](float* p, float v, size_t n) {
    fill_f32_kernel<<<(unsigned)((n + 255) / 256), 256, 0, stream>>>(p, v, n);
  };
  auto fillu = [&](unsigned* p, unsigned v, size_t n) {
    fill_u32_kernel<<<(unsigned)((n + 255) / 256), 256, 0, stream>>>(p, v, n);
  };
  auto gemm = [&](const float* A, const float* W, const float* bias, float* Cc,
                  int M, int K, int Ncol) {
    int Mt = (M + 15) / 16, Nt = (Ncol + 15) / 16, Kt = (K + 31) / 32;
    size_t ta = (size_t)Mt * Kt * 32;
    pack_frag_kernel<<<(unsigned)((ta + 255) / 256), 256, 0, stream>>>(A, Apack, M, K, K, 1, Kt, ta);
    size_t tb = (size_t)Nt * Kt * 32;
    pack_frag_kernel<<<(unsigned)((tb + 255) / 256), 256, 0, stream>>>(W, Bpack, Ncol, K, 1, Ncol, Kt, tb);
    dim3 g((Mt + 3) / 4, (Nt + 7) / 8);
    wmma_gemm_kernel<<<g, 256, 0, stream>>>(Apack, Bpack, bias, Cc, M, Ncol, Kt);
  };
  auto bnfun = [&](const float* X, const float* res, const float* g, const float* be,
                   float* Y, int M, int C, int relu) {
    bn_stats_kernel<<<C, 256, 0, stream>>>(X, M, C, mu, rstd);
    size_t tot = (size_t)M * C;
    bn_apply_kernel<<<(unsigned)((tot + 255) / 256), 256, 0, stream>>>(X, res, mu, rstd, g, be, Y, M, C, relu);
  };
  auto gat = [&](const float* hin, int ci, int co, int base, float* gmean) {
    int HC = Hh * co;
    gemm(hin, fp(base + 0), nullptr, xl, Nn, ci, HC);   // Wl
    gemm(hin, fp(base + 1), nullptr, xr, Nn, ci, HC);   // Wr
    fillu(mx, 0u, (size_t)Nn * Hh);
    fillf(den, 0.f, (size_t)Nn * Hh);
    fillf(acc, 0.f, (size_t)Nn * HC);
    size_t waves = (size_t)E2 * Hh, thr = waves * 32;
    edge_pass1_kernel<<<(unsigned)((thr + 255) / 256), 256, 0, stream>>>(
        src2, dst2, ea2, fp(base + 2), xl, xr, fp(base + 3), logits, mx, E2, Hh, co);
    edge_pass2_kernel<<<(unsigned)((waves + 255) / 256), 256, 0, stream>>>(
        dst2, mx, logits, den, E2, Hh);
    edge_pass3_kernel<<<(unsigned)((thr + 255) / 256), 256, 0, stream>>>(
        src2, dst2, logits, den, xl, acc, E2, Hh, co);
    head_mean_kernel<<<(unsigned)(((size_t)Nn * co + 255) / 256), 256, 0, stream>>>(
        acc, fp(base + 4), gmean, Nn, Hh, co);
  };
  auto run_head = [&](int base, const int* idx, int n, float fill, float* outseg) {
    pair_build_kernel<<<(unsigned)((n + 255) / 256), 256, 0, stream>>>(latent, idx, idx + n, fill, Pbuf, n);
    gemm(Pbuf, fp(base + 0), fp(base + 1), Ubuf, n, 65, 256);
    bnfun(Ubuf, nullptr, fp(base + 2), fp(base + 3), Vbuf, n, 256, 1);
    float* cur = Vbuf; float* oth = Ubuf;
    for (int r = 0; r < 3; ++r) {
      int rb = base + 4 + r * 4;
      gemm(cur, fp(rb + 0), fp(rb + 1), oth, n, 256, 256);
      bnfun(oth, cur, fp(rb + 2), fp(rb + 3), oth, n, 256, 1);   // relu(bn(..)+res)
      float* t = cur; cur = oth; oth = t;
    }
    gemm(cur, fp(base + 16), fp(base + 17), oth, n, 256, 32);
    bnfun(oth, nullptr, fp(base + 18), fp(base + 19), oth, n, 32, 1);
    fc6_kernel<<<(unsigned)(((size_t)n * 32 + 255) / 256), 256, 0, stream>>>(
        oth, fp(base + 20), fp(base + 21), outseg, n);
  };

  // -------- self-loop attr (fill_value='mean') --------
  const int* eidx = ip(1);
  const float* eattr = fp(2);
  fillf(cnt, 0.f, (size_t)Nn);
  fillf(easum, 0.f, (size_t)Nn * 16);
  loop_accum_kernel<<<(Ee + 255) / 256, 256, 0, stream>>>(eidx + Ee, eattr, cnt, easum, Ee);
  build_edges_kernel<<<(E2 + 255) / 256, 256, 0, stream>>>(eidx, eidx + Ee, eattr, cnt, easum,
                                                           src2, dst2, ea2, Ee, Nn);

  // -------- trunk --------
  bnfun(fp(0), nullptr, fp(7), fp(8), hA, Nn, Fdim, 0);      // bn0
  gat(hA, Fdim, 256, 9, gm);  bnfun(gm, nullptr, fp(14), fp(15), hB, Nn, 256, 1);  // g1/bn1
  gat(hB, 256, 256, 16, gm);  bnfun(gm, hB, fp(21), fp(22), hA, Nn, 256, 1);       // g2/bn2 +res
  gat(hA, 256, 256, 23, gm);  bnfun(gm, hA, fp(28), fp(29), hB, Nn, 256, 1);       // g3/bn3 +res
  gat(hB, 256, 256, 30, gm);  bnfun(gm, hB, fp(35), fp(36), hA, Nn, 256, 1);       // g4/bn4 +res
  gat(hA, 256, 32, 37, gm);   bnfun(gm, nullptr, fp(42), fp(43), latent, Nn, 32, 1); // g5/bn5

  // -------- pair heads --------
  float* outp = (float*)d_out;
  size_t o1 = (size_t)Nn * 32;
  run_head(44,  ip(3), NB,  1.0f, outp + o1);
  run_head(66,  ip(4), NA,  2.0f, outp + o1 + NB);
  run_head(88,  ip(5), ND,  3.0f, outp + o1 + NB + NA);
  run_head(110, ip(6), NH4, 4.0f, outp + o1 + NB + NA + ND);
}
